// MultiHeadAttention_Linear_11312943857747
// MI455X (gfx1250) — compile-verified, matrix operand order checked by reading, not run on hardware
//
#include <hip/hip_runtime.h>
#include <hip/hip_bf16.h>

// ---------------------------------------------------------------------------
// Linear attention block for MI455X (gfx1250), wave32 + WMMA 16x16x32 f16.
// B=4, S=4096, DM=1024, H=16, HD=64
// All GEMM LDS tiles use a unified fragment-friendly layout T[row][64] halves
// (A row-major, B transposed), so every WMMA fragment is two aligned
// ds_load_b128 ops.
// ---------------------------------------------------------------------------

#define B_   4
#define S_   4096
#define DM_  1024
#define H_   16
#define HD_  64
#define ROWS_ (B_ * S_)          // 16384
#define NELEM_ (ROWS_ * DM_)     // 16777216
#define KT_   64                 // K tile per stage

typedef __attribute__((ext_vector_type(16))) _Float16 v16h;
typedef __attribute__((ext_vector_type(8)))  _Float16 v8h;
typedef __attribute__((ext_vector_type(4)))  _Float16 v4h;
typedef __attribute__((ext_vector_type(8)))  float    v8f;

// ---------------- WMMA fragment loader (from LDS, layout T[r][64]) ---------
// For A: r = M row.  For B (stored transposed): r = N col.
// lane: r = r0 + (lane&15); K-half group = (lane>>4)*8 within the kk subtile.
__device__ __forceinline__ v16h frag(const _Float16* T, int r0, int kk, int lane) {
    const int r  = r0 + (lane & 15);
    const int kb = ((lane >> 4) << 3) + kk;
    const v8h* p = (const v8h*)(T + r * KT_ + kb);   // 16B aligned
    const v8h lo = p[0];   // K = kb .. kb+7
    const v8h hi = p[2];   // K = kb+16 .. kb+23
    return __builtin_shufflevector(lo, hi, 0, 1, 2, 3, 4, 5, 6, 7,
                                   8, 9, 10, 11, 12, 13, 14, 15);
}

#define WMMA_F16(A, Bm, C) \
    __builtin_amdgcn_wmma_f32_16x16x32_f16(false, (A), false, (Bm), (short)0, (C), false, false)

// 8 WMMAs over a 64-deep K tile for one wave's 32x32 quadrant.
#define GEMM_STEP(As, Bt, wm, wn, lane)                                   \
    _Pragma("unroll")                                                     \
    for (int kk = 0; kk < KT_; kk += 32) {                                \
        v16h a0 = frag((As), (wm) + 0,  kk, (lane));                      \
        v16h a1 = frag((As), (wm) + 16, kk, (lane));                      \
        v16h b0 = frag((Bt), (wn) + 0,  kk, (lane));                      \
        v16h b1 = frag((Bt), (wn) + 16, kk, (lane));                      \
        c00 = WMMA_F16(a0, b0, c00);                                      \
        c01 = WMMA_F16(a0, b1, c01);                                      \
        c10 = WMMA_F16(a1, b0, c10);                                      \
        c11 = WMMA_F16(a1, b1, c11);                                      \
    }

// ---------------- elementwise converts -------------------------------------
__global__ void cvt_f32_f16_kernel(const float* __restrict__ src,
                                   _Float16* __restrict__ dst, int n4) {
    int i = blockIdx.x * blockDim.x + threadIdx.x;
    if (i >= n4) return;
    const float4 v = ((const float4*)src)[i];
    v4h o;
    o[0] = (_Float16)v.x; o[1] = (_Float16)v.y;
    o[2] = (_Float16)v.z; o[3] = (_Float16)v.w;
    ((v4h*)dst)[i] = o;
}

// wt[w][k][n] = W_w[n][k]  (transpose + f16), w in {q,k,v}
__global__ void cvt_w_kernel(const float* __restrict__ Wq,
                             const float* __restrict__ Wk,
                             const float* __restrict__ Wv,
                             _Float16* __restrict__ wt) {
    int i = blockIdx.x * blockDim.x + threadIdx.x;   // 0 .. 3*1024*1024-1
    if (i >= 3 * DM_ * DM_) return;
    int w  = i >> 20;
    int kk = (i >> 10) & (DM_ - 1);
    int nn = i & (DM_ - 1);
    const float* W = (w == 0) ? Wq : ((w == 1) ? Wk : Wv);
    wt[i] = (_Float16)W[nn * DM_ + kk];
}

// ---------------- QKV GEMM: out = xh @ wt + bias ---------------------------
// grid (ROWS/64, DM/64); block 128 (4 waves); each wave a 32x32 quadrant.
// do_softmax=1: fused per-row softmax over the 64-wide (head-aligned) tile.
__global__ __launch_bounds__(128)
void gemm_qkv_kernel(const _Float16* __restrict__ xh,   // ROWS x DM
                     const _Float16* __restrict__ wt,   // DM x DM (K x N)
                     const float* __restrict__ bias,    // DM
                     _Float16* __restrict__ out,        // ROWS x DM
                     int do_softmax) {
    __shared__ _Float16 As[64 * KT_];   // As[m][k]
    __shared__ _Float16 Bt[64 * KT_];   // Bt[n][k]  (transposed B)
    __shared__ float    Cs[64 * 65];

    const int tid  = threadIdx.x;
    const int lane = tid & 31;
    const int wid  = tid >> 5;
    const int m0   = blockIdx.x * 64;
    const int n0   = blockIdx.y * 64;
    const int wm   = (wid >> 1) * 32;
    const int wn   = (wid & 1) * 32;

    v8f c00 = {}, c01 = {}, c10 = {}, c11 = {};

    const int row  = tid >> 1;          // 0..63
    const int cblk = (tid & 1) * 32;    // 0 or 32

    for (int kt = 0; kt < DM_; kt += KT_) {
        {   // A tile 64x64: vector load + vector LDS store
            const _Float16* src = xh + (size_t)(m0 + row) * DM_ + kt + cblk;
            if (kt + KT_ < DM_) __builtin_prefetch(src + KT_, 0, 1);
            v8h* dst = (v8h*)(As + row * KT_ + cblk);
#pragma unroll
            for (int j = 0; j < 4; ++j) dst[j] = ((const v8h*)src)[j];
        }
        {   // B tile 64x64 -> transposed into Bt[n][k]
            const _Float16* src = wt + (size_t)(kt + row) * DM_ + n0 + cblk;
            v8h t[4];
#pragma unroll
            for (int j = 0; j < 4; ++j) t[j] = ((const v8h*)src)[j];
#pragma unroll
            for (int j = 0; j < 32; ++j)
                Bt[(cblk + j) * KT_ + row] = t[j >> 3][j & 7];
        }
        __syncthreads();
        GEMM_STEP(As, Bt, wm, wn, lane);
        __syncthreads();
    }

    // stage accumulators (+bias) to LDS
    {
        const int nlo = lane & 15;
        const int rhi = (lane >> 4) << 3;
#pragma unroll
        for (int mi = 0; mi < 2; ++mi)
#pragma unroll
            for (int ni = 0; ni < 2; ++ni) {
                const v8f acc = (mi == 0) ? (ni == 0 ? c00 : c01)
                                          : (ni == 0 ? c10 : c11);
                const int col = wn + ni * 16 + nlo;
                const float bv = bias[n0 + col];
#pragma unroll
                for (int r = 0; r < 8; ++r) {
                    const int rr = wm + mi * 16 + r + rhi;
                    Cs[rr * 65 + col] = acc[r] + bv;
                }
            }
    }
    __syncthreads();

    if (do_softmax) {
        if (tid < 64) {
            const float* crow = &Cs[tid * 65];
            float mx = crow[0];
            for (int j = 1; j < 64; ++j) mx = fmaxf(mx, crow[j]);
            float s = 0.f;
            for (int j = 0; j < 64; ++j) s += __expf(crow[j] - mx);
            const float inv = 1.0f / s;
            _Float16* dst = out + (size_t)(m0 + tid) * DM_ + n0;
            for (int j = 0; j < 64; ++j)
                dst[j] = (_Float16)(__expf(crow[j] - mx) * inv);
        }
    } else {
        _Float16* dst = out + (size_t)(m0 + row) * DM_ + n0 + cblk;
        const float* srcr = &Cs[row * 65 + cblk];
#pragma unroll
        for (int j = 0; j < 32; ++j) dst[j] = (_Float16)srcr[j];
    }
}

// ---------------- K softmax over seq (axis -2), in place -------------------
// grid B*DM blocks; block 256; column of 4096 elements (stride DM).
__global__ __launch_bounds__(256)
void ksoftmax_kernel(_Float16* __restrict__ kh) {
    __shared__ float red[256];
    const int c = blockIdx.x & (DM_ - 1);
    const int b = blockIdx.x >> 10;
    const size_t base = (size_t)b * S_ * DM_ + c;
    const int tid = threadIdx.x;

    float vals[16];
    float mx = -3.0e38f;
#pragma unroll
    for (int i = 0; i < 16; ++i) {
        const int s = tid + i * 256;
        vals[i] = (float)kh[base + (size_t)s * DM_];
        mx = fmaxf(mx, vals[i]);
    }
    red[tid] = mx; __syncthreads();
    for (int off = 128; off > 0; off >>= 1) {
        if (tid < off) red[tid] = fmaxf(red[tid], red[tid + off]);
        __syncthreads();
    }
    mx = red[0]; __syncthreads();

    float s = 0.f;
#pragma unroll
    for (int i = 0; i < 16; ++i) { vals[i] = __expf(vals[i] - mx); s += vals[i]; }
    red[tid] = s; __syncthreads();
    for (int off = 128; off > 0; off >>= 1) {
        if (tid < off) red[tid] += red[tid + off];
        __syncthreads();
    }
    const float inv = 1.0f / red[0];
#pragma unroll
    for (int i = 0; i < 16; ++i) {
        const int ss = tid + i * 256;
        kh[base + (size_t)ss * DM_] = (_Float16)(vals[i] * inv);
    }
}

// ---------------- gmap[bh] = k_head^T @ v_head  (64x64, K=S=4096) ----------
// grid B*H; block 128 (4 waves, 32x32 quadrants).
__global__ __launch_bounds__(128)
void gemm_gmap_kernel(const _Float16* __restrict__ kh,
                      const _Float16* __restrict__ vh,
                      _Float16* __restrict__ gmap) {     // [B*H][64][64]
    __shared__ _Float16 As[64 * KT_];   // As[d][s]
    __shared__ _Float16 Bt[64 * KT_];   // Bt[e][s]
    const int bh = blockIdx.x;
    const int b = bh >> 4, h = bh & 15;
    const size_t base = (size_t)b * S_ * DM_ + h * HD_;
    const int tid = threadIdx.x, lane = tid & 31, wid = tid >> 5;
    const int wm = (wid >> 1) * 32, wn = (wid & 1) * 32;

    v8f c00 = {}, c01 = {}, c10 = {}, c11 = {};

    const int srow = tid >> 1;          // s within tile: 0..63
    const int cblk = (tid & 1) * 32;    // head-dim block: 0 or 32

    for (int st = 0; st < S_; st += KT_) {
        const _Float16* ksrc = kh + base + (size_t)(st + srow) * DM_ + cblk;
        const _Float16* vsrc = vh + base + (size_t)(st + srow) * DM_ + cblk;
        if (st + KT_ < S_) __builtin_prefetch(ksrc + (size_t)KT_ * DM_, 0, 1);
        v8h tk[4], tv[4];
#pragma unroll
        for (int j = 0; j < 4; ++j) { tk[j] = ((const v8h*)ksrc)[j];
                                      tv[j] = ((const v8h*)vsrc)[j]; }
#pragma unroll
        for (int j = 0; j < 32; ++j) {   // scatter transpose into LDS
            As[(cblk + j) * KT_ + srow] = tk[j >> 3][j & 7];
            Bt[(cblk + j) * KT_ + srow] = tv[j >> 3][j & 7];
        }
        __syncthreads();
        GEMM_STEP(As, Bt, wm, wn, lane);
        __syncthreads();
    }

    _Float16* dst = gmap + (size_t)bh * (HD_ * HD_);
    const int nlo = lane & 15;
    const int rhi = (lane >> 4) << 3;
#pragma unroll
    for (int mi = 0; mi < 2; ++mi)
#pragma unroll
        for (int ni = 0; ni < 2; ++ni) {
            const v8f acc = (mi == 0) ? (ni == 0 ? c00 : c01)
                                      : (ni == 0 ? c10 : c11);
            const int e = wn + ni * 16 + nlo;
#pragma unroll
            for (int r = 0; r < 8; ++r) {
                const int d = wm + mi * 16 + r + rhi;
                dst[d * HD_ + e] = (_Float16)acc[r];
            }
        }
}

// ---------------- o = q_head @ gmap[bh]  -> written into d_out -------------
// grid (S/64, B*H); block 128; tile 64x64, K=HD=64 (single K tile).
__global__ __launch_bounds__(128)
void gemm_o_kernel(const _Float16* __restrict__ qh,
                   const _Float16* __restrict__ gmap,
                   float* __restrict__ out) {
    __shared__ _Float16 As[64 * KT_];   // As[m][k]
    __shared__ _Float16 Bt[64 * KT_];   // Bt[n][k]
    const int s0 = blockIdx.x * 64;
    const int bh = blockIdx.y;
    const int b = bh >> 4, h = bh & 15;
    const int tid = threadIdx.x, lane = tid & 31, wid = tid >> 5;
    const int wm = (wid >> 1) * 32, wn = (wid & 1) * 32;

    v8f c00 = {}, c01 = {}, c10 = {}, c11 = {};

    const int row  = tid >> 1;
    const int cblk = (tid & 1) * 32;

    {   // A: As[m][k] = qh[b, s0+m, h*64+k]   (vector copy)
        const _Float16* src =
            qh + ((size_t)(b * S_ + s0 + row)) * DM_ + h * HD_ + cblk;
        v8h* dst = (v8h*)(As + row * KT_ + cblk);
#pragma unroll
        for (int j = 0; j < 4; ++j) dst[j] = ((const v8h*)src)[j];
    }
    {   // B: Bt[n][k] = gmap[bh][k][n]  (transpose scatter)
        const _Float16* src =
            gmap + (size_t)bh * (HD_ * HD_) + (size_t)row * HD_ + cblk;
        v8h t[4];
#pragma unroll
        for (int j = 0; j < 4; ++j) t[j] = ((const v8h*)src)[j];
#pragma unroll
        for (int j = 0; j < 32; ++j)
            Bt[(cblk + j) * KT_ + row] = t[j >> 3][j & 7];
    }
    __syncthreads();
    GEMM_STEP(As, Bt, wm, wn, lane);

    const int nlo = lane & 15;
    const int rhi = (lane >> 4) << 3;
#pragma unroll
    for (int mi = 0; mi < 2; ++mi)
#pragma unroll
        for (int ni = 0; ni < 2; ++ni) {
            const v8f acc = (mi == 0) ? (ni == 0 ? c00 : c01)
                                      : (ni == 0 ? c10 : c11);
            const int n = wn + ni * 16 + nlo;
#pragma unroll
            for (int r = 0; r < 8; ++r) {
                const int m = wm + mi * 16 + r + rhi;
                out[((size_t)(b * S_ + s0 + m)) * DM_ + h * HD_ + n] = acc[r];
            }
        }
}

// ---------------- y = LN(x + o), in place on d_out -------------------------
__global__ __launch_bounds__(256)
void layernorm_kernel(const float* __restrict__ x,
                      const float* __restrict__ gamma,
                      const float* __restrict__ beta,
                      float* __restrict__ out) {
    __shared__ float rs_[256];
    __shared__ float rq_[256];
    const int tid = threadIdx.x;
    const size_t base = (size_t)blockIdx.x * DM_;

    float y[4];
    float s = 0.f, q = 0.f;
#pragma unroll
    for (int i = 0; i < 4; ++i) {
        const int c = tid + i * 256;
        const float v = x[base + c] + out[base + c];
        y[i] = v; s += v; q += v * v;
    }
    rs_[tid] = s; rq_[tid] = q; __syncthreads();
    for (int off = 128; off > 0; off >>= 1) {
        if (tid < off) { rs_[tid] += rs_[tid + off]; rq_[tid] += rq_[tid + off]; }
        __syncthreads();
    }
    const float mu  = rs_[0] * (1.0f / DM_);
    const float var = rq_[0] * (1.0f / DM_) - mu * mu;
    const float r   = rsqrtf(var + 1e-5f);
#pragma unroll
    for (int i = 0; i < 4; ++i) {
        const int c = tid + i * 256;
        out[base + c] = (y[i] - mu) * r * gamma[c] + beta[c];
    }
}

// ---------------------------------------------------------------------------
extern "C" void kernel_launch(void* const* d_in, const int* in_sizes, int n_in,
                              void* d_out, int out_size, void* d_ws, size_t ws_size,
                              hipStream_t stream) {
    const float* x     = (const float*)d_in[0];
    // d_in[1] mask, d_in[2] pad_mask: unused by the reference math
    const float* Wq    = (const float*)d_in[3];
    const float* bq    = (const float*)d_in[4];
    const float* Wk    = (const float*)d_in[5];
    const float* bk    = (const float*)d_in[6];
    const float* Wv    = (const float*)d_in[7];
    const float* bv    = (const float*)d_in[8];
    const float* gamma = (const float*)d_in[9];
    const float* beta  = (const float*)d_in[10];
    float* out = (float*)d_out;

    char* ws = (char*)d_ws;
    size_t off = 0;
    _Float16* xh  = (_Float16*)(ws + off); off += (size_t)NELEM_ * 2;          // 32 MB
    _Float16* wt  = (_Float16*)(ws + off); off += (size_t)3 * DM_ * DM_ * 2;   // 6 MB
    _Float16* qh  = (_Float16*)(ws + off); off += (size_t)NELEM_ * 2;          // 32 MB
    _Float16* kh  = (_Float16*)(ws + off); off += (size_t)NELEM_ * 2;          // 32 MB
    _Float16* vh  = (_Float16*)(ws + off); off += (size_t)NELEM_ * 2;          // 32 MB
    _Float16* gmp = (_Float16*)(ws + off); off += (size_t)B_ * H_ * HD_ * HD_ * 2;

    // 1) convert x -> f16 (vectorized 4-wide)
    cvt_f32_f16_kernel<<<(NELEM_ / 4) / 256, 256, 0, stream>>>(x, xh, NELEM_ / 4);
    // 2) convert + transpose weights -> f16 (K x N)
    cvt_w_kernel<<<(3 * DM_ * DM_) / 256, 256, 0, stream>>>(Wq, Wk, Wv, wt);

    // 3) Q/K/V projections (WMMA); Q gets fused head-dim softmax
    dim3 gg(ROWS_ / 64, DM_ / 64);
    gemm_qkv_kernel<<<gg, 128, 0, stream>>>(xh, wt,                 bq, qh, 1);
    gemm_qkv_kernel<<<gg, 128, 0, stream>>>(xh, wt + DM_ * DM_,     bk, kh, 0);
    gemm_qkv_kernel<<<gg, 128, 0, stream>>>(xh, wt + 2 * DM_ * DM_, bv, vh, 0);

    // 4) K softmax over seq (in place)
    ksoftmax_kernel<<<B_ * DM_, 256, 0, stream>>>(kh);

    // 5) global map: k^T v per head (WMMA, K=4096)
    gemm_gmap_kernel<<<B_ * H_, 128, 0, stream>>>(kh, vh, gmp);

    // 6) o = q @ gmap -> d_out (WMMA)
    dim3 go(S_ / 64, B_ * H_);
    gemm_o_kernel<<<go, 128, 0, stream>>>(qh, gmp, out);

    // 7) y = LayerNorm(x + o) in place on d_out
    layernorm_kernel<<<ROWS_, 256, 0, stream>>>(x, gamma, beta, out);
}
